// DecoderCell_34385508172375
// MI455X (gfx1250) — compile-verified, matrix operand order checked by reading, not run on hardware
//
#include <hip/hip_runtime.h>

// ---------- types ----------
typedef __attribute__((ext_vector_type(16))) __bf16 v16bf;
typedef __attribute__((ext_vector_type(8)))  float  v8f;
typedef __attribute__((ext_vector_type(4)))  unsigned int v4u;
typedef __attribute__((ext_vector_type(2)))  unsigned int v2u;

union FragU { v4u q[2]; v16bf f; };   // 32 bytes = 16 bf16 (one WMMA A/B fragment per lane)

// f32 -> bf16 (round to nearest even), packed helpers
__device__ __forceinline__ unsigned int bf16_1(float f) {
  unsigned int u = __float_as_uint(f);
  return (u + 0x7FFFu + ((u >> 16) & 1u)) >> 16;
}
__device__ __forceinline__ unsigned int bf16_pk(float lo, float hi) {
  return bf16_1(lo) | (bf16_1(hi) << 16);
}

// problem sizes
#define LQ   512
#define BN   256
#define QD   512
#define IND  512
#define HD   1024
#define AD   512
#define NROW (LQ*BN)        // 131072 flattened (l,b) rows
#define KIH  (IND + QD + HD) // 2048 = W_ih inner dim
#define KX   3584           // row length; x concat h0 fills IN + Q + H + H = 3072
#define N4H  4096
#define APAD 8
#define ASTR (QD + APAD)    // LDS row stride in ushorts

// ---------- kernel 1: att1_w query-half -> bf16 (row-major A x Q) ----------
__global__ void k_wq_bf16(const float* __restrict__ att1_w, unsigned short* __restrict__ wq) {
  int i = blockIdx.x * 256 + threadIdx.x;          // 0 .. 512*512-1
  int a = i >> 9, q = i & 511;
  wq[i] = (unsigned short)bf16_1(att1_w[a * (QD + HD) + HD + q]);
}

// ---------- kernel 1b: combined [W_ih | W_hh] -> bf16 rows of length KX ----------
__global__ __launch_bounds__(256) void k_wc_bf16(const float* __restrict__ Wih,
                                                 const float* __restrict__ Whh,
                                                 unsigned short* __restrict__ wc) {
  int n = blockIdx.x, t = threadIdx.x;             // n = 0..4095
  const float4* s0 = (const float4*)(Wih + (size_t)n * KIH);
  const float4* s1 = (const float4*)(Whh + (size_t)n * HD);
  v2u* dst = (v2u*)(wc + (size_t)n * KX);
  for (int j = t; j < KIH / 4; j += 256) {         // 512 float4s
    float4 v = s0[j];
    v2u p; p.x = bf16_pk(v.x, v.y); p.y = bf16_pk(v.z, v.w);
    dst[j] = p;
  }
  for (int j = t; j < HD / 4; j += 256) {          // 256 float4s
    float4 v = s1[j];
    v2u p; p.x = bf16_pk(v.x, v.y); p.y = bf16_pk(v.z, v.w);
    dst[KIH / 4 + j] = p;
  }
}

// ---------- kernel 2: u0[b,a] = h0[b] . att1_w[a, 0:H] + att1_b[a] ----------
__global__ __launch_bounds__(256) void k_u0(const float* __restrict__ h0,
                                            const float* __restrict__ att1_w,
                                            const float* __restrict__ att1_b,
                                            float* __restrict__ u0) {
  int b  = threadIdx.x;
  int a0 = blockIdx.x * 16;
  float acc[16] = {};
  for (int k = 0; k < HD; ++k) {
    float hv = h0[b * HD + k];
#pragma unroll
    for (int j = 0; j < 16; ++j)
      acc[j] = fmaf(hv, att1_w[(a0 + j) * (QD + HD) + k], acc[j]);
  }
#pragma unroll
  for (int j = 0; j < 16; ++j)
    u0[b * AD + a0 + j] = acc[j] + att1_b[a0 + j];
}

// ---------- kernel 3: big attention GEMM + fused tanh/dot epilogue ----------
// block: 256 thr (8 waves). tile M=32 rows x N=512 (all of A). wave: 16 rows x 128 cols.
// att_raw[r] = att2_b + sum_a att2_w[a] * tanh(u0[b,a] + query_row(r) . Wq[a,:])
__global__ __launch_bounds__(256) void k_att_gemm(const float* __restrict__ qdata,
                                                  const unsigned short* __restrict__ wq,
                                                  const float* __restrict__ u0,
                                                  const float* __restrict__ w2,
                                                  const float* __restrict__ b2,
                                                  float* __restrict__ att_raw) {
  __shared__ unsigned short sA[32 * ASTR];   // 32 x 512 bf16 query tile (padded)
  __shared__ float sRed[32][4];

  const int tid = threadIdx.x;
  const int wave = tid >> 5, lane = tid & 31;
  const int rowBase = blockIdx.x * 32;

  // stage query tile: f32 -> bf16 into LDS (coalesced float4 loads)
#pragma unroll
  for (int i = 0; i < 16; ++i) {
    int e = tid + i * 256;                   // float4 index within 32x512 tile
    int row = e >> 7, c4 = e & 127;
    float4 v = ((const float4*)qdata)[(size_t)(rowBase + row) * 128 + c4];
    v2u p; p.x = bf16_pk(v.x, v.y); p.y = bf16_pk(v.z, v.w);
    *(v2u*)&sA[row * ASTR + c4 * 4] = p;
  }
  __syncthreads();

  const int mSub  = (wave & 1) * 16;         // which 16-row half of the 32-row tile
  const int nBase = (wave >> 1) * 128;       // 128-wide column group
  const int lm = lane & 15;                  // A: row within fragment / B: column n
  const int lh = lane >> 4;                  // K-half selector

  v8f acc[8] = {};
  for (int kk = 0; kk < QD; kk += 32) {
    FragU a;
    const unsigned short* ap = &sA[(mSub + lm) * ASTR + kk + lh * 8];
    a.q[0] = *(const v4u*)ap;                // K = kk + lh*8 .. +7
    a.q[1] = *(const v4u*)(ap + 16);         // K = kk + 16 + lh*8 .. +7
#pragma unroll
    for (int t = 0; t < 8; ++t) {
      FragU bf;
      const unsigned short* bp = &wq[(size_t)(nBase + t * 16 + lm) * QD + kk + lh * 8];
      bf.q[0] = *(const v4u*)bp;
      bf.q[1] = *(const v4u*)(bp + 16);
      acc[t] = __builtin_amdgcn_wmma_f32_16x16x32_bf16(false, a.f, false, bf.f,
                                                       (short)0, acc[t], false, false);
    }
  }

  // epilogue: hid = acc + u0;  rowsum += w2[a]*tanh(hid)
  const int mLoc0 = lh * 8;                  // C/D: vgpr v -> local row mLoc0+v, col lm
  float rowsum[8] = {};
#pragma unroll
  for (int t = 0; t < 8; ++t) {
    int aIdx = nBase + t * 16 + lm;
    float w2v = w2[aIdx];
#pragma unroll
    for (int v = 0; v < 8; ++v) {
      int r = rowBase + mSub + mLoc0 + v;
      int b = r & (BN - 1);
      float hid = acc[t][v] + u0[b * AD + aIdx];
      rowsum[v] = fmaf(w2v, tanhf(hid), rowsum[v]);
    }
  }
  // reduce the 16 columns held across each 16-lane half
#pragma unroll
  for (int v = 0; v < 8; ++v) {
    float s = rowsum[v];
    s += __shfl_xor(s, 1, 16);
    s += __shfl_xor(s, 2, 16);
    s += __shfl_xor(s, 4, 16);
    s += __shfl_xor(s, 8, 16);
    rowsum[v] = s;
  }
  if (lm == 0) {
#pragma unroll
    for (int v = 0; v < 8; ++v)
      sRed[mSub + mLoc0 + v][wave >> 1] = rowsum[v];
  }
  __syncthreads();
  if (tid < 32) {
    float s = sRed[tid][0] + sRed[tid][1] + sRed[tid][2] + sRed[tid][3] + b2[0];
    att_raw[rowBase + tid] = s;
  }
}

// ---------- kernel 4: masked softmax over L_q (max before mask, as reference) ----------
__global__ __launch_bounds__(256) void k_softmax(const float* __restrict__ att_raw,
                                                 const float* __restrict__ qmask,
                                                 float* __restrict__ wts) {
  __shared__ float red[256];
  int b = blockIdx.x, t = threadIdx.x;
  float v0 = att_raw[t * BN + b];
  float v1 = att_raw[(t + 256) * BN + b];
  red[t] = fmaxf(v0, v1);
  __syncthreads();
  for (int s = 128; s > 0; s >>= 1) { if (t < s) red[t] = fmaxf(red[t], red[t + s]); __syncthreads(); }
  float mx = red[0];
  __syncthreads();
  float e0 = __expf(v0 - mx) * qmask[t * BN + b];
  float e1 = __expf(v1 - mx) * qmask[(t + 256) * BN + b];
  red[t] = e0 + e1;
  __syncthreads();
  for (int s = 128; s > 0; s >>= 1) { if (t < s) red[t] += red[t + s]; __syncthreads(); }
  float inv = 1.f / red[0];
  wts[t * BN + b] = e0 * inv;
  wts[(t + 256) * BN + b] = e1 * inv;
}

// ---------- kernel 5: ctx[b,q] = sum_l w[l,b]*query[l,b,q] (second query pass) ----------
__global__ __launch_bounds__(256) void k_ctx(const float* __restrict__ qdata,
                                             const float* __restrict__ wts,
                                             float* __restrict__ out_ctx) {
  int b = blockIdx.x, t = threadIdx.x;       // t -> float2 of q
  float2 acc = make_float2(0.f, 0.f);
  for (int l = 0; l < LQ; ++l) {
    float w = wts[l * BN + b];
    float2 q = ((const float2*)qdata)[((size_t)l * BN + b) * 256 + t];
    acc.x = fmaf(w, q.x, acc.x);
    acc.y = fmaf(w, q.y, acc.y);
  }
  ((float2*)out_ctx)[b * 256 + t] = acc;
}

// ---------- kernel 6: build xh = bf16[input | ctx | history[parent] | h0] ----------
__global__ __launch_bounds__(256) void k_build_x(const float* __restrict__ input,
                                                 const float* __restrict__ ctx,
                                                 const int* __restrict__ pidx,
                                                 const float* __restrict__ history,
                                                 const float* __restrict__ h0,
                                                 unsigned short* __restrict__ xh) {
  int b = blockIdx.x, t = threadIdx.x;
  int p = pidx[b];
  unsigned short* row = xh + (size_t)b * KX;
  for (int j = t; j < IND; j += 256)  row[j]        = (unsigned short)bf16_1(input[b * IND + j]);
  for (int j = t; j < QD;  j += 256)  row[IND + j]  = (unsigned short)bf16_1(ctx[b * QD + j]);
  for (int j = t; j < HD;  j += 256)  row[IND + QD + j] =
      (unsigned short)bf16_1(history[((size_t)p * BN + b) * HD + j]);
  for (int j = t; j < HD;  j += 256)  row[IND + QD + HD + j] =
      (unsigned short)bf16_1(h0[b * HD + j]);
}

// ---------- kernel 7: LSTM GEMM gates = [x,h0] @ wc^T + biases ----------
// M=256, N=4096, K=3584. block tile M=32 x N=512 (grid 8x8). A and B both bf16 from global.
__global__ __launch_bounds__(256) void k_lstm_gemm(const unsigned short* __restrict__ xh,
                                                   const unsigned short* __restrict__ wc,
                                                   const float* __restrict__ bih,
                                                   const float* __restrict__ bhh,
                                                   float* __restrict__ gates) {
  const int tid = threadIdx.x, wave = tid >> 5, lane = tid & 31;
  const int lm = lane & 15, lh = lane >> 4;
  const int rowBase = blockIdx.y * 32;
  const int mSub = (wave & 1) * 16;
  const int nBase = blockIdx.x * 512 + (wave >> 1) * 128;

  v8f acc[8] = {};
  const unsigned short* arow = xh + (size_t)(rowBase + mSub + lm) * KX;

  for (int kk = 0; kk < KX; kk += 32) {
    FragU a;
    a.q[0] = *(const v4u*)(arow + kk + lh * 8);
    a.q[1] = *(const v4u*)(arow + kk + 16 + lh * 8);
#pragma unroll
    for (int t = 0; t < 8; ++t) {
      const unsigned short* bp = wc + (size_t)(nBase + t * 16 + lm) * KX + kk + lh * 8;
      FragU bf;
      bf.q[0] = *(const v4u*)bp;
      bf.q[1] = *(const v4u*)(bp + 16);
      acc[t] = __builtin_amdgcn_wmma_f32_16x16x32_bf16(false, a.f, false, bf.f,
                                                       (short)0, acc[t], false, false);
    }
  }
#pragma unroll
  for (int t = 0; t < 8; ++t) {
    int n = nBase + t * 16 + lm;
    float bias = bih[n] + bhh[n];
#pragma unroll
    for (int v = 0; v < 8; ++v) {
      int m = lh * 8 + v;
      gates[(size_t)(rowBase + mSub + m) * N4H + n] = acc[t][v] + bias;
    }
  }
}

// ---------- kernel 8: LSTM nonlinearities -> h1, c1 ----------
__global__ __launch_bounds__(256) void k_lstm_ep(const float* __restrict__ gates,
                                                 const float* __restrict__ c0,
                                                 float* __restrict__ out) {
  int idx = blockIdx.x * 256 + threadIdx.x;   // 0 .. 256*1024-1
  int b = idx >> 10, j = idx & (HD - 1);
  const float* g = gates + (size_t)b * N4H;
  float gi = g[j], gf = g[j + HD], gg = g[j + 2 * HD], go = g[j + 3 * HD];
  float si = 1.f / (1.f + __expf(-gi));
  float sf = 1.f / (1.f + __expf(-gf));
  float so = 1.f / (1.f + __expf(-go));
  float c1 = sf * c0[idx] + si * tanhf(gg);
  float h1 = so * tanhf(c1);
  out[BN * QD + idx] = h1;                     // h1 after ctx block
  out[BN * QD + BN * HD + idx] = c1;           // c1 after h1 block
}

// ---------- host side ----------
extern "C" void kernel_launch(void* const* d_in, const int* in_sizes, int n_in,
                              void* d_out, int out_size, void* d_ws, size_t ws_size,
                              hipStream_t stream) {
  const float* qdata  = (const float*)d_in[0];
  const float* qmask  = (const float*)d_in[1];
  const float* input  = (const float*)d_in[2];
  const int*   pidx   = (const int*)  d_in[3];
  const float* hist   = (const float*)d_in[4];
  const float* h0     = (const float*)d_in[5];
  const float* c0     = (const float*)d_in[6];
  const float* Wih    = (const float*)d_in[7];
  const float* Whh    = (const float*)d_in[8];
  const float* bih    = (const float*)d_in[9];
  const float* bhh    = (const float*)d_in[10];
  const float* att1_w = (const float*)d_in[11];
  const float* att1_b = (const float*)d_in[12];
  const float* att2_w = (const float*)d_in[13];
  const float* att2_b = (const float*)d_in[14];
  float* out = (float*)d_out;

  char* ws = (char*)d_ws;
  unsigned short* wq      = (unsigned short*)(ws);                  // 512 KB
  float*          u0      = (float*)(ws + 524288);                  // 512 KB
  float*          att_raw = (float*)(ws + 1048576);                 // 512 KB
  float*          wts     = (float*)(ws + 1572864);                 // 512 KB
  unsigned short* xh      = (unsigned short*)(ws + 2097152);        // 1.75 MB
  float*          gates   = (float*)(ws + 3932160);                 // 4 MB
  unsigned short* wc      = (unsigned short*)(ws + 8126464);        // 28 MB (4096 x 3584 bf16)

  k_wq_bf16 <<<1024, 256, 0, stream>>>(att1_w, wq);
  k_wc_bf16 <<<N4H,  256, 0, stream>>>(Wih, Whh, wc);
  k_u0      <<<32,   256, 0, stream>>>(h0, att1_w, att1_b, u0);
  k_att_gemm<<<NROW / 32, 256, 0, stream>>>(qdata, wq, u0, att2_w, att2_b, att_raw);
  k_softmax <<<BN,   256, 0, stream>>>(att_raw, qmask, wts);
  k_ctx     <<<BN,   256, 0, stream>>>(qdata, wts, out);
  k_build_x <<<BN,   256, 0, stream>>>(input, out, pidx, hist, h0, xh);
  dim3 g(8, 8);
  k_lstm_gemm<<<g,   256, 0, stream>>>(xh, wc, bih, bhh, gates);
  k_lstm_ep <<<BN * HD / 256, 256, 0, stream>>>(gates, c0, out);
}
